// RegNet_37005438222411
// MI455X (gfx1250) — matrix operand order repaired and checked by reading, not compiled, not run on hardware
//
#include <hip/hip_runtime.h>

#define H_DIM 256
#define DIN   128

// ---------------- CDNA5 WMMA types ----------------
typedef __bf16 v16bf __attribute__((ext_vector_type(16)));
typedef float  v8f   __attribute__((ext_vector_type(8)));

union FragBF {
    v16bf v;
    uint4 q[2];          // two 16-byte halves (8 bf16 each)
};

__device__ __forceinline__ unsigned short f32_to_bf16_rne(float f) {
    unsigned int u = __float_as_uint(f);
    unsigned int r = u + 0x7FFFu + ((u >> 16) & 1u);   // round-to-nearest-even
    return (unsigned short)(r >> 16);
}

// ---------------------------------------------------------------------------
// GEMM: out[n,H] = in[n,K] @ W[K,H] (+ bias). bf16 WMMA, f32 accumulate.
// Block = 256 threads = 8 waves; block tile = 128 rows x 64 cols.
// Wave w owns rows [w*16, w*16+16); 4 WMMA C fragments across the 64 cols.
// ---------------------------------------------------------------------------
__global__ __launch_bounds__(256) void gemm_wmma_bf16(
    const float* __restrict__ in, const float* __restrict__ W,
    const float* __restrict__ bias, float* __restrict__ out,
    int n, int K, int useBias)
{
    // padded row stride 40 ushorts = 80B (16B aligned, spreads LDS banks)
    __shared__ __align__(16) unsigned short Atile[128][40]; // [row][k]  bf16
    __shared__ __align__(16) unsigned short Btile[64][40];  // [col][k]  bf16 (transposed)

    const int tid   = threadIdx.x;
    const int wave  = tid >> 5;
    const int lane  = tid & 31;
    const int l16   = lane & 15;
    const int hi    = lane >> 4;          // 0: lanes 0-15, 1: lanes 16-31
    const int mBase = blockIdx.x * 128;
    const int nBase = blockIdx.y * 64;

    v8f acc[4];
#pragma unroll
    for (int f = 0; f < 4; ++f)
#pragma unroll
        for (int j = 0; j < 8; ++j) acc[f][j] = 0.0f;

    for (int k0 = 0; k0 < K; k0 += 32) {
        // ---- stage A tile: 128 x 32 f32 -> bf16 (1024 float4 slots) ----
#pragma unroll
        for (int i = 0; i < 4; ++i) {
            int sid = tid + 256 * i;
            int row = sid >> 3;           // 8 float4 per row
            int kq  = sid & 7;
            int gr  = mBase + row;
            float4 vq = make_float4(0.f, 0.f, 0.f, 0.f);
            if (gr < n)
                vq = *(const float4*)(in + (size_t)gr * K + k0 + kq * 4);
            unsigned short* p = &Atile[row][kq * 4];
            p[0] = f32_to_bf16_rne(vq.x); p[1] = f32_to_bf16_rne(vq.y);
            p[2] = f32_to_bf16_rne(vq.z); p[3] = f32_to_bf16_rne(vq.w);
        }
        // ---- stage B tile transposed: 32 x 64 f32 -> Btile[col][k] ----
#pragma unroll
        for (int i = 0; i < 2; ++i) {
            int sid  = tid + 256 * i;
            int krow = sid >> 4;          // 16 float4 per k-row
            int nq   = sid & 15;
            const float4 wq = *(const float4*)(W + (size_t)(k0 + krow) * H_DIM + nBase + nq * 4);
            Btile[nq * 4 + 0][krow] = f32_to_bf16_rne(wq.x);
            Btile[nq * 4 + 1][krow] = f32_to_bf16_rne(wq.y);
            Btile[nq * 4 + 2][krow] = f32_to_bf16_rne(wq.z);
            Btile[nq * 4 + 3][krow] = f32_to_bf16_rne(wq.w);
        }
        __syncthreads();

        // ---- A fragment (16x32 bf16 ISA layout) ----
        // lane<16:  K0..7 in half0, K16..23 in half1
        // lane>=16: K8..15 in half0, K24..31 in half1
        FragBF a;
        const unsigned short* arow = &Atile[wave * 16 + l16][0];
        a.q[0] = *(const uint4*)(arow + hi * 8);
        a.q[1] = *(const uint4*)(arow + 16 + hi * 8);

#pragma unroll
        for (int f = 0; f < 4; ++f) {
            // B fragment (32x16 bf16): lane holds col=l16, same K order as A:
            // K0..7,K16..23 (lanes 0-15) or K8..15,K24..31 (lanes 16-31)
            FragBF b;
            const unsigned short* brow = &Btile[f * 16 + l16][0];
            b.q[0] = *(const uint4*)(brow + hi * 8);
            b.q[1] = *(const uint4*)(brow + 16 + hi * 8);
            acc[f] = __builtin_amdgcn_wmma_f32_16x16x32_bf16(
                false, a.v, false, b.v, (short)0, acc[f], false, false);
        }
        __syncthreads();
    }

    // ---- store C: VGPR r -> M = r (lanes 0-15) / 8+r (lanes 16-31), N = l16
#pragma unroll
    for (int f = 0; f < 4; ++f) {
        const int col = nBase + f * 16 + l16;
        const float bv = useBias ? bias[col] : 0.0f;
#pragma unroll
        for (int r = 0; r < 8; ++r) {
            int row = mBase + wave * 16 + r + hi * 8;
            if (row < n)
                out[(size_t)row * H_DIM + col] = acc[f][r] + bv;
        }
    }
}

// ---------------------------------------------------------------------------
// Degree / symmetric-norm kernels: deg = indegree(dst) + 1 (self loop),
// dinv = rsqrt(deg).
// ---------------------------------------------------------------------------
__global__ void deg_init(float* dinv, int n) {
    int i = blockIdx.x * 256 + threadIdx.x;
    if (i < n) dinv[i] = 1.0f;                       // self-loop contribution
}
__global__ void deg_edges(const int* __restrict__ dst, float* dinv, int e) {
    int i = blockIdx.x * 256 + threadIdx.x;
    if (i < e) atomicAdd(&dinv[dst[i]], 1.0f);
}
__global__ void deg_finalize(float* dinv, int n) {
    int i = blockIdx.x * 256 + threadIdx.x;
    if (i < n) dinv[i] = rsqrtf(dinv[i]);            // deg >= 1 always
}

// out[i,:] = h[i,:] * dinv[i]^2   (self-loop term, also zero-inits the target)
__global__ void selfloop_init(const float* __restrict__ h,
                              const float* __restrict__ dinv,
                              float* __restrict__ out, int total) {
    int g = blockIdx.x * 256 + threadIdx.x;
    if (g < total) {
        int row = g >> 8;                            // H_DIM == 256
        float di = dinv[row];
        out[g] = h[g] * di * di;
    }
}

// edge scatter: out[dst,:] += h[src,:] * dinv[src]*dinv[dst]
// one block of 256 threads handles 8 edges; thread t = feature t.
// edge indices are threadIdx-uniform -> scalar (SMEM) loads.
__global__ __launch_bounds__(256) void scatter_edges(
    const float* __restrict__ h, float* __restrict__ out,
    const int* __restrict__ src, const int* __restrict__ dst,
    const float* __restrict__ dinv, int nEdges)
{
    const int t  = threadIdx.x;
    const int e0 = blockIdx.x * 8;
#pragma unroll
    for (int i = 0; i < 8; ++i) {
        int e = e0 + i;
        if (e >= nEdges) return;
        int s = src[e];
        int d = dst[e];
        float nrm = dinv[s] * dinv[d];
        atomicAdd(&out[(size_t)d * H_DIM + t], h[(size_t)s * H_DIM + t] * nrm);
    }
}

// h = relu(h + b)
__global__ void bias_relu(float* __restrict__ h, const float* __restrict__ b, int total) {
    int g = blockIdx.x * 256 + threadIdx.x;
    if (g < total) h[g] = fmaxf(h[g] + b[g & (H_DIM - 1)], 0.0f);
}

// decoder: y[i] = dot(h[i,:], dec_W) + dec_b ; one wave32 per node
__global__ __launch_bounds__(256) void decoder_dot(
    const float* __restrict__ h, const float* __restrict__ dw,
    const float* __restrict__ db, float* __restrict__ y, int n)
{
    const int lane = threadIdx.x & 31;
    const int wid  = threadIdx.x >> 5;
    const int node = blockIdx.x * 8 + wid;
    if (node >= n) return;
    const float* hp = h + (size_t)node * H_DIM;
    float s = 0.0f;
#pragma unroll
    for (int j = 0; j < 8; ++j) {
        int k = lane + 32 * j;
        s += hp[k] * dw[k];
    }
#pragma unroll
    for (int off = 16; off > 0; off >>= 1)
        s += __shfl_xor(s, off, 32);
    if (lane == 0) y[node] = s + db[0];
}

// ---------------------------------------------------------------------------
extern "C" void kernel_launch(void* const* d_in, const int* in_sizes, int n_in,
                              void* d_out, int out_size, void* d_ws, size_t ws_size,
                              hipStream_t stream) {
    const float* x     = (const float*)d_in[0];
    const int*   ei    = (const int*)  d_in[1];
    const float* enc_W = (const float*)d_in[2];
    const float* enc_b = (const float*)d_in[3];
    const float* Wc[3] = {(const float*)d_in[4], (const float*)d_in[6], (const float*)d_in[8]};
    const float* bc[3] = {(const float*)d_in[5], (const float*)d_in[7], (const float*)d_in[9]};
    const float* dec_W = (const float*)d_in[10];
    const float* dec_b = (const float*)d_in[11];
    float*       y     = (float*)d_out;

    const int N = in_sizes[0] / DIN;     // 50000
    const int E = in_sizes[1] / 2;       // 800000
    const int* src = ei;
    const int* dst = ei + E;

    // workspace: hA | hB | dinv  (~103 MB)
    float* hA   = (float*)d_ws;
    float* hB   = hA + (size_t)N * H_DIM;
    float* dinv = hB + (size_t)N * H_DIM;

    const int total   = N * H_DIM;
    const dim3 blk(256);
    const dim3 gN((N + 255) / 256);
    const dim3 gE((E + 255) / 256);
    const dim3 gT((total + 255) / 256);
    const dim3 gScat((E + 7) / 8);
    const dim3 gGemm((N + 127) / 128, H_DIM / 64);
    const dim3 gDec((N + 7) / 8);

    // symmetric normalization coefficients
    deg_init    <<<gN, blk, 0, stream>>>(dinv, N);
    deg_edges   <<<gE, blk, 0, stream>>>(dst, dinv, E);
    deg_finalize<<<gN, blk, 0, stream>>>(dinv, N);

    // encoder: hA = x @ enc_W + enc_b
    gemm_wmma_bf16<<<gGemm, blk, 0, stream>>>(x, enc_W, enc_b, hA, N, DIN, 1);

    // three GCN convs, each followed by ReLU
    for (int l = 0; l < 3; ++l) {
        gemm_wmma_bf16<<<gGemm, blk, 0, stream>>>(hA, Wc[l], nullptr, hB, N, H_DIM, 0);
        selfloop_init <<<gT, blk, 0, stream>>>(hB, dinv, hA, total);
        scatter_edges <<<gScat, blk, 0, stream>>>(hB, hA, src, dst, dinv, E);
        bias_relu     <<<gT, blk, 0, stream>>>(hA, bc[l], total);
    }

    // decoder
    decoder_dot<<<gDec, blk, 0, stream>>>(hA, dec_W, dec_b, y, N);
}
